// mixed_decayRNN_fix_bump_72464688218246
// MI455X (gfx1250) — matrix operand order repaired and checked by reading, not compiled, not run on hardware
//
#include <hip/hip_runtime.h>

// Problem constants (match reference)
#define ALPHA_C   0.1f
#define BB        256
#define TT        200
#define IN_DIM    33
#define HH        2048
#define OUTD      4
#define KPAD      64              // padded input-projection K
#define KROW      (HH + KPAD)     // 2112: concatenated K row length
#define NSTEPS    (KROW / 32)     // 66 K-steps

// Tiling
#define BM 64
#define BN 128
#define BK 32
#define LDW 40            // padded LDS row width (bf16 elems) -> 80B stride, 16B aligned

typedef __attribute__((ext_vector_type(16))) __bf16 v16bf;
typedef __attribute__((ext_vector_type(8)))  float  v8f;

union FragAB { uint4 q[2]; v16bf v; };
union AccU   { v8f v; float f[8]; };

// fp32 -> bf16 bits, round-half-up (cheap, branch-free)
__device__ __forceinline__ unsigned bf16_rhu(float a) {
    return (__float_as_uint(a) + 0x8000u) >> 16;
}
__device__ __forceinline__ unsigned pack2_bf16(float a, float b) {
    unsigned ua = (__float_as_uint(a) + 0x8000u) >> 16;
    unsigned ub = (__float_as_uint(b) + 0x8000u) & 0xFFFF0000u;
    return ua | ub;
}

// ---------------- prologue kernels (run once per launch) ----------------

// wcat[H][KROW] = [Wrec row (2048) | Wi2h row (33) | zeros] in bf16
__global__ __launch_bounds__(256)
void build_wcat_kernel(const float* __restrict__ wrec,
                       const float* __restrict__ wi2h,
                       unsigned short* __restrict__ o) {
    int row = blockIdx.x;                       // 0..H-1
    for (int q = threadIdx.x; q < KROW / 4; q += 256) {
        int j = q * 4;
        float v[4];
        #pragma unroll
        for (int c = 0; c < 4; ++c) {
            int jc = j + c;
            float f = 0.f;
            if (jc < HH)                 f = wrec[(size_t)row * HH + jc];
            else if (jc < HH + IN_DIM)   f = wi2h[(size_t)row * IN_DIM + (jc - HH)];
            v[c] = f;
        }
        *(uint2*)((void*)(o + (size_t)row * KROW + j)) =
            make_uint2(pack2_bf16(v[0], v[1]), pack2_bf16(v[2], v[3]));
    }
}

// x fp32 [B*T,IN] -> bf16 [B*T,KPAD] zero-padded
__global__ __launch_bounds__(256)
void pad_x_kernel(const float* __restrict__ x, unsigned short* __restrict__ o) {
    int e   = blockIdx.x * 256 + threadIdx.x;   // B*T*16 total
    int row = e >> 4;
    int j   = (e & 15) * 4;
    float v[4];
    #pragma unroll
    for (int c = 0; c < 4; ++c) {
        int jc = j + c;
        v[c] = (jc < IN_DIM) ? x[(size_t)row * IN_DIM + jc] : 0.f;
    }
    *(uint2*)((void*)(o + (size_t)row * KPAD + j)) =
        make_uint2(pack2_bf16(v[0], v[1]), pack2_bf16(v[2], v[3]));
}

// zero a contiguous uint4 span (hb1 + fp32 zero row)
__global__ __launch_bounds__(256)
void zero_ws_kernel(uint4* __restrict__ p) {
    p[(size_t)blockIdx.x * 256 + threadIdx.x] = make_uint4(0u, 0u, 0u, 0u);
}

// seed x_0 into the tail of hb1 rows
__global__ __launch_bounds__(256)
void seed_x0_kernel(const unsigned short* __restrict__ x_bf,
                    unsigned short* __restrict__ hb1) {
    int e   = blockIdx.x * 256 + threadIdx.x;   // B*8 total
    int row = e >> 3;
    int q   = (e & 7) * 8;
    *(uint4*)((void*)&hb1[(size_t)row * KROW + HH + q]) =
        *(const uint4*)((const void*)&x_bf[((size_t)row * TT + 0) * KPAD + q]);
}

// ---------------- WMMA core ----------------

__device__ __forceinline__ void mma_tile(const unsigned short* __restrict__ Alds,
                                         const unsigned short* __restrict__ Blds,
                                         int lane, int wave_m, int wave_n,
                                         v8f acc[2][2]) {
    const int rlane = lane & 15;
    const int hi    = lane & 16;
    FragAB afrag[2], bfrag[2];
    #pragma unroll
    for (int mi = 0; mi < 2; ++mi) {
        int row = wave_m * 32 + mi * 16 + rlane;
        int kb  = hi ? 8 : 0;            // lanes 16-31: K=8..15 / 24..31
        afrag[mi].q[0] = *(const uint4*)((const void*)&Alds[row * LDW + kb]);
        afrag[mi].q[1] = *(const uint4*)((const void*)&Alds[row * LDW + kb + 16]);
    }
    #pragma unroll
    for (int ni = 0; ni < 2; ++ni) {
        int row = wave_n * 32 + ni * 16 + rlane;
        int kb  = hi ? 8 : 0;            // lanes 16-31: K=8..15 / 24..31
        bfrag[ni].q[0] = *(const uint4*)((const void*)&Blds[row * LDW + kb]);
        bfrag[ni].q[1] = *(const uint4*)((const void*)&Blds[row * LDW + kb + 16]);
    }
    #pragma unroll
    for (int mi = 0; mi < 2; ++mi)
        #pragma unroll
        for (int ni = 0; ni < 2; ++ni)
            acc[mi][ni] = __builtin_amdgcn_wmma_f32_16x16x32_bf16(
                false, afrag[mi].v, false, bfrag[ni].v,
                (short)0, acc[mi][ni], false, false);
}

// One recurrent timestep. Rows of hb_* are [h (2048) | x_t (64)] bf16,
// rows of wcat are [Wrec | Wi2h | 0] bf16 -> single uniform 66-step K loop.
__global__ __launch_bounds__(256)
void rnn_step_kernel(const unsigned short* __restrict__ wcat,     // [H,KROW]
                     const unsigned short* __restrict__ x_bf,     // [B*T,KPAD]
                     const unsigned short* __restrict__ hb_read,  // [B,KROW]
                     unsigned short* __restrict__ hb_write,       // [B,KROW]
                     const float* __restrict__ hprev_base,        // fp32 h_{t-1} (or zeros)
                     size_t hprev_stride,                         // row stride (0 at t==0)
                     const float* __restrict__ noise,             // [T,B,H]
                     float* __restrict__ hiddens,                 // [B,T,H] (d_out)
                     int t)
{
    __shared__ unsigned short Alds[2][BM * LDW];
    __shared__ unsigned short Blds[2][BN * LDW];

    const int tid     = threadIdx.x;
    const int lane    = tid & 31;
    const int wid     = tid >> 5;
    const int wave_m  = wid & 1;
    const int wave_n  = wid >> 1;
    const int block_m = blockIdx.y * BM;
    const int block_n = blockIdx.x * BN;

    // staging: each thread moves one uint4 (8 bf16) of A, two of B per K-step
    const int rowA  = tid >> 2;          // 0..63
    const int quadA = (tid & 3) * 8;
    int rowB[2], quadB[2];
    #pragma unroll
    for (int g = 0; g < 2; ++g) {
        int e = tid + g * 256;
        rowB[g]  = e >> 2;               // 0..127
        quadB[g] = (e & 3) * 8;
    }

    const unsigned short* aPtr = hb_read + (size_t)(block_m + rowA) * KROW + quadA;
    const unsigned short* bPtr0 = wcat + (size_t)(block_n + rowB[0]) * KROW + quadB[0];
    const unsigned short* bPtr1 = wcat + (size_t)(block_n + rowB[1]) * KROW + quadB[1];
    const int aoff = rowA * LDW + quadA;
    const int boff0 = rowB[0] * LDW + quadB[0];
    const int boff1 = rowB[1] * LDW + quadB[1];

    v8f acc[2][2];
    {
        v8f z = {};
        #pragma unroll
        for (int i = 0; i < 2; ++i)
            #pragma unroll
            for (int j = 0; j < 2; ++j) acc[i][j] = z;
    }

    // ---- double-buffered pipeline: tile s in buf[s&1]; 1 barrier / K-step ----
    uint4 ra, rb0, rb1;
    // tile 0 -> buf0
    ra  = *(const uint4*)((const void*)aPtr);
    rb0 = *(const uint4*)((const void*)bPtr0);
    rb1 = *(const uint4*)((const void*)bPtr1);
    *(uint4*)((void*)&Alds[0][aoff])  = ra;
    *(uint4*)((void*)&Blds[0][boff0]) = rb0;
    *(uint4*)((void*)&Blds[0][boff1]) = rb1;
    // fetch tile 1
    ra  = *(const uint4*)((const void*)(aPtr + BK));
    rb0 = *(const uint4*)((const void*)(bPtr0 + BK));
    rb1 = *(const uint4*)((const void*)(bPtr1 + BK));

    for (int s = 0; s < NSTEPS; ++s) {
        __syncthreads();
        if (s < NSTEPS - 1) {            // commit tile s+1 into the other buffer
            int b = (s + 1) & 1;
            *(uint4*)((void*)&Alds[b][aoff])  = ra;
            *(uint4*)((void*)&Blds[b][boff0]) = rb0;
            *(uint4*)((void*)&Blds[b][boff1]) = rb1;
        }
        if (s < NSTEPS - 2) {            // fetch tile s+2
            int o = (s + 2) * BK;
            ra  = *(const uint4*)((const void*)(aPtr + o));
            rb0 = *(const uint4*)((const void*)(bPtr0 + o));
            rb1 = *(const uint4*)((const void*)(bPtr1 + o));
        }
        mma_tile(Alds[s & 1], Blds[s & 1], lane, wave_m, wave_n, acc);
    }

    // ---- epilogue: noise + relu + alpha-decay; fp32 h_t + bf16 copy ----
    const int rlane = lane & 15;
    const int mhalf = (lane & 16) ? 8 : 0;   // C/D layout: lanes 16-31 -> M = 8+r
    #pragma unroll
    for (int mi = 0; mi < 2; ++mi) {
        #pragma unroll
        for (int ni = 0; ni < 2; ++ni) {
            AccU u; u.v = acc[mi][ni];
            #pragma unroll
            for (int r = 0; r < 8; ++r) {
                int m = block_m + wave_m * 32 + mi * 16 + mhalf + r;
                int n = block_n + wave_n * 32 + ni * 16 + rlane;
                float pre = u.f[r] + noise[((size_t)t * BB + m) * (size_t)HH + n];
                float hp  = hprev_base[(size_t)m * hprev_stride + n];
                float h   = fmaxf(pre, 0.f) * ALPHA_C + (1.f - ALPHA_C) * hp;
                hiddens[((size_t)m * TT + t) * (size_t)HH + n] = h;
                hb_write[(size_t)m * KROW + n] = (unsigned short)bf16_rhu(h);
            }
        }
    }

    // blocks in the n==0 column also stage x_{t+1} into hb_write row tails
    if (blockIdx.x == 0 && t + 1 < TT) {
        #pragma unroll
        for (int g = 0; g < 2; ++g) {
            int e   = tid + g * 256;     // 0..511
            int row = e >> 3;            // 0..63
            int q   = (e & 7) * 8;
            *(uint4*)((void*)&hb_write[(size_t)(block_m + row) * KROW + HH + q]) =
                *(const uint4*)((const void*)
                    &x_bf[((size_t)(block_m + row) * TT + (t + 1)) * KPAD + q]);
        }
    }
}

// Output head: one wave per (b,t) row; 4 dots of length H, wave32 reduction, softmax.
__global__ __launch_bounds__(256)
void rnn_out_kernel(const float* __restrict__ hiddens, // [B,T,H]
                    const float* __restrict__ Wh2o,    // [OUT,H]
                    float* __restrict__ outs)          // [B,T,OUT]
{
    const int lane = threadIdx.x & 31;
    const int wid  = threadIdx.x >> 5;
    const int row  = blockIdx.x * 8 + wid;     // row index over B*T
    const float* hrow = hiddens + (size_t)row * HH;

    float a0 = 0.f, a1 = 0.f, a2 = 0.f, a3 = 0.f;
    for (int j = lane; j < HH; j += 32) {
        float hv = hrow[j];
        a0 = fmaf(hv, Wh2o[0 * HH + j], a0);
        a1 = fmaf(hv, Wh2o[1 * HH + j], a1);
        a2 = fmaf(hv, Wh2o[2 * HH + j], a2);
        a3 = fmaf(hv, Wh2o[3 * HH + j], a3);
    }
    #pragma unroll
    for (int off = 16; off > 0; off >>= 1) {
        a0 += __shfl_xor(a0, off, 32);
        a1 += __shfl_xor(a1, off, 32);
        a2 += __shfl_xor(a2, off, 32);
        a3 += __shfl_xor(a3, off, 32);
    }
    if (lane == 0) {
        float mx = fmaxf(fmaxf(a0, a1), fmaxf(a2, a3));
        float e0 = expf(a0 - mx), e1 = expf(a1 - mx);
        float e2 = expf(a2 - mx), e3 = expf(a3 - mx);
        float s  = e0 + e1 + e2 + e3;
        float inv = 1.f / s;
        outs[(size_t)row * OUTD + 0] = e0 * inv;
        outs[(size_t)row * OUTD + 1] = e1 * inv;
        outs[(size_t)row * OUTD + 2] = e2 * inv;
        outs[(size_t)row * OUTD + 3] = e3 * inv;
    }
}

extern "C" void kernel_launch(void* const* d_in, const int* in_sizes, int n_in,
                              void* d_out, int out_size, void* d_ws, size_t ws_size,
                              hipStream_t stream) {
    (void)in_sizes; (void)n_in; (void)out_size; (void)ws_size;
    const float* x     = (const float*)d_in[0];
    const float* noise = (const float*)d_in[1];
    const float* Wi2h  = (const float*)d_in[2];
    const float* Wrec  = (const float*)d_in[3];
    const float* Wh2o  = (const float*)d_in[4];

    float* hiddens = (float*)d_out;                       // [B,T,H]
    float* outs    = hiddens + (size_t)BB * TT * HH;      // [B,T,OUT]

    // workspace layout (~17.5 MB)
    unsigned short* wcat = (unsigned short*)d_ws;                  // H*KROW
    unsigned short* x_bf = wcat + (size_t)HH * KROW;               // B*T*KPAD
    unsigned short* hb0  = x_bf + (size_t)BB * TT * KPAD;          // B*KROW
    unsigned short* hb1  = hb0  + (size_t)BB * KROW;               // B*KROW
    float*          zrow = (float*)(hb1 + (size_t)BB * KROW);      // HH zeros (fp32)

    // one-time prologue
    build_wcat_kernel<<<HH, 256, 0, stream>>>(Wrec, Wi2h, wcat);
    pad_x_kernel<<<(BB * TT * 16) / 256, 256, 0, stream>>>(x, x_bf);
    // zero hb1 (t=0 reads) + fp32 zero row, contiguous span
    {
        size_t bytes = (size_t)BB * KROW * 2 + (size_t)HH * 4;
        zero_ws_kernel<<<(unsigned)(bytes / 16 / 256), 256, 0, stream>>>((uint4*)hb1);
    }
    seed_x0_kernel<<<(BB * 8) / 256, 256, 0, stream>>>(x_bf, hb1);

    dim3 grid(HH / BN, BB / BM);   // (16, 4) = 64 workgroups per step
    for (int t = 0; t < TT; ++t) {
        const unsigned short* hb_read  = (t & 1) ? hb0 : hb1;  // (t-1) parity
        unsigned short*       hb_write = (t & 1) ? hb1 : hb0;  // t parity
        const float* hprev_base  = (t > 0) ? (hiddens + (size_t)(t - 1) * HH) : zrow;
        size_t       hprev_stride = (t > 0) ? (size_t)TT * HH : 0;
        rnn_step_kernel<<<grid, 256, 0, stream>>>(
            wcat, x_bf, hb_read, hb_write, hprev_base, hprev_stride,
            noise, hiddens, t);
    }

    rnn_out_kernel<<<(BB * TT) / 8, 256, 0, stream>>>(hiddens, Wh2o, outs);
}
